// RNN_31782757990499
// MI455X (gfx1250) — matrix operand order repaired and checked by reading, not compiled, not run on hardware
//
#include <hip/hip_runtime.h>

// ---------- types ----------
typedef __attribute__((ext_vector_type(16))) __bf16          v16bf;
typedef __attribute__((ext_vector_type(8)))  float           v8f;
typedef __attribute__((ext_vector_type(8)))  unsigned short  u16x8;

union Frag16 {           // 16 bf16 = 32 bytes, loaded as two b128 chunks
    v16bf bf;
    u16x8 h[2];
};

__device__ __forceinline__ unsigned short f32_to_bf16(float f) {
    union { float f; unsigned int u; } v; v.f = f;
    unsigned int r = v.u + 0x7FFFu + ((v.u >> 16) & 1u);   // round-nearest-even
    return (unsigned short)(r >> 16);
}
__device__ __forceinline__ float bf16_to_f32(unsigned short u) {
    union { unsigned int i; float f; } v; v.i = ((unsigned int)u) << 16;
    return v.f;
}

// A-fragment (16x32 bf16): lane(lo,hi) holds K chunks {kb+hi*8..+7} and {kb+16+hi*8..+7}
__device__ __forceinline__ void load_afrag(Frag16& f, const unsigned short* row, int kb, int hi) {
    f.h[0] = *(const u16x8*)(row + kb + hi * 8);
    f.h[1] = *(const u16x8*)(row + kb + 16 + hi * 8);
}
// B-fragment (32x16 bf16): lane n holds K chunks {kb+hi*8..+7} and {kb+16+hi*8..+7}
__device__ __forceinline__ void load_bfrag(Frag16& f, const unsigned short* row, int kb, int hi) {
    f.h[0] = *(const u16x8*)(row + kb + hi * 8);
    f.h[1] = *(const u16x8*)(row + kb + 16 + hi * 8);
}
__device__ __forceinline__ v8f wmma_bf16(const Frag16& a, const Frag16& b, v8f c) {
    return __builtin_amdgcn_wmma_f32_16x16x32_bf16(
               false, a.bf, false, b.bf, (short)0, c, false, false);
}

// ---------- elementwise convert f32 -> bf16 (4 elems / thread) ----------
__global__ void cvt_f32_bf16_kernel(const float* __restrict__ src,
                                    unsigned short* __restrict__ dst, int n4) {
    int i = (blockIdx.x * blockDim.x + threadIdx.x);
    if (i < n4) {
        const float4 f = ((const float4*)src)[i];
        ushort4 o;
        o.x = f32_to_bf16(f.x); o.y = f32_to_bf16(f.y);
        o.z = f32_to_bf16(f.z); o.w = f32_to_bf16(f.w);
        ((ushort4*)dst)[i] = o;
    }
}

__global__ void zero_bf16_kernel(unsigned short* __restrict__ dst, int n) {
    int i = blockIdx.x * blockDim.x + threadIdx.x;
    if (i < n) dst[i] = 0;
}

// ---------- big GEMM: out[m,n] = bf16( sum_k A[m,k]*Wt[n,k] + bias[n] ) ----------
// Register-blocked 32(M) x 64(N) per wave: 2 A-frags x 4 B-frags -> 8 WMMA / k-step.
// Block = 128 threads = 4 waves; wave w -> nBase = (blockIdx.y*4+w)*64; mBase = blockIdx.x*32.
__global__ __launch_bounds__(128)
void gemm_xw_kernel(const unsigned short* __restrict__ A,    // [M,K] bf16
                    const unsigned short* __restrict__ Wt,   // [N,K] bf16
                    const float* __restrict__ bias,          // [N]
                    unsigned short* __restrict__ out,        // [M,N] bf16
                    int N, int K) {
    const int lane  = threadIdx.x & 31;
    const int wave  = threadIdx.x >> 5;
    const int lo    = lane & 15;
    const int hi    = lane >> 4;
    const int mBase = blockIdx.x * 32;
    const int nBase = (blockIdx.y * 4 + wave) * 64;

    const unsigned short* arow[2];
    const unsigned short* brow[4];
    #pragma unroll
    for (int i = 0; i < 2; ++i) arow[i] = A  + (size_t)(mBase + i * 16 + lo) * K;
    #pragma unroll
    for (int j = 0; j < 4; ++j) brow[j] = Wt + (size_t)(nBase + j * 16 + lo) * K;

    v8f acc[2][4] = {};
    #pragma unroll 2
    for (int kb = 0; kb < K; kb += 32) {
        Frag16 a[2], b[4];
        #pragma unroll
        for (int i = 0; i < 2; ++i) load_afrag(a[i], arow[i], kb, hi);
        #pragma unroll
        for (int j = 0; j < 4; ++j) load_bfrag(b[j], brow[j], kb, hi);
        #pragma unroll
        for (int i = 0; i < 2; ++i)
            #pragma unroll
            for (int j = 0; j < 4; ++j)
                acc[i][j] = wmma_bf16(a[i], b[j], acc[i][j]);
    }
    #pragma unroll
    for (int j = 0; j < 4; ++j) {
        const int n  = nBase + j * 16 + lo;
        const float bn = bias[n];
        #pragma unroll
        for (int i = 0; i < 2; ++i) {
            #pragma unroll
            for (int r = 0; r < 8; ++r) {
                const int m = mBase + i * 16 + r + 8 * hi;   // C layout: VGPR r -> M=r+8*hi
                out[(size_t)m * N + n] = f32_to_bf16(acc[i][j][r] + bn);
            }
        }
    }
}

// ---------- fused M=64 GEMM: v = Hin@Wt^T + bias (+ xw[:,t,:]) ; relu ; store ----------
// M-blocked 64(M) x 16(N) per wave: the streamed operand (U/V1 row panel) is loaded
// ONCE per k-step and reused by 4 WMMAs -> minimal (compulsory) weight traffic.
// Grid: blockIdx.x in [0, N/64); wave w -> nTile = blockIdx.x*4 + w.
__global__ __launch_bounds__(128)
void gemm64_relu_kernel(const unsigned short* __restrict__ Hin,   // [64,K] bf16
                        const unsigned short* __restrict__ Wt,    // [N,K] bf16
                        const float* __restrict__ bias,           // [N]
                        const unsigned short* __restrict__ xw,    // [64,T,N] bf16 or null
                        int t, int T,
                        unsigned short* __restrict__ HoutBf,      // [64,N] bf16 or null
                        float* __restrict__ outF,                 // [64,N] f32 or null
                        int N, int K) {
    const int lane  = threadIdx.x & 31;
    const int wave  = threadIdx.x >> 5;
    const int lo    = lane & 15;
    const int hi    = lane >> 4;
    const int nBase = (blockIdx.x * 4 + wave) * 16;

    const unsigned short* arow[4];
    #pragma unroll
    for (int i = 0; i < 4; ++i) arow[i] = Hin + (size_t)(i * 16 + lo) * K;
    const unsigned short* brow = Wt + (size_t)(nBase + lo) * K;

    v8f acc[4] = {};
    #pragma unroll 2
    for (int kb = 0; kb < K; kb += 32) {
        Frag16 a[4], b;
        load_bfrag(b, brow, kb, hi);
        #pragma unroll
        for (int i = 0; i < 4; ++i) load_afrag(a[i], arow[i], kb, hi);
        #pragma unroll
        for (int i = 0; i < 4; ++i) acc[i] = wmma_bf16(a[i], b, acc[i]);
    }
    const int n  = nBase + lo;
    const float bn = bias[n];
    #pragma unroll
    for (int i = 0; i < 4; ++i) {
        #pragma unroll
        for (int r = 0; r < 8; ++r) {
            const int m = i * 16 + r + 8 * hi;
            float v = acc[i][r] + bn;
            if (xw) v += bf16_to_f32(xw[((size_t)m * T + t) * N + n]);
            v = v > 0.0f ? v : 0.0f;                       // relu
            if (HoutBf) HoutBf[(size_t)m * N + n] = f32_to_bf16(v);
            else        outF  [(size_t)m * N + n] = v;
        }
    }
}

// ---------- tiny head: z2 = relu(z1@V2^T + V2_b); out = log_softmax(z2) ----------
__global__ __launch_bounds__(256)
void head_kernel(const float* __restrict__ z1,      // [64,H] f32
                 const float* __restrict__ V2w,     // [5,H] f32
                 const float* __restrict__ V2b,     // [5]
                 float* __restrict__ out,           // [64,5]
                 int H) {
    __shared__ float red[5][256];
    const int b = blockIdx.x, tid = threadIdx.x;
    float loc[5] = {0.f, 0.f, 0.f, 0.f, 0.f};
    for (int k = tid; k < H; k += 256) {
        const float x = z1[(size_t)b * H + k];
        #pragma unroll
        for (int c = 0; c < 5; ++c) loc[c] += x * V2w[(size_t)c * H + k];
    }
    #pragma unroll
    for (int c = 0; c < 5; ++c) red[c][tid] = loc[c];
    __syncthreads();
    for (int s = 128; s > 0; s >>= 1) {
        if (tid < s) {
            #pragma unroll
            for (int c = 0; c < 5; ++c) red[c][tid] += red[c][tid + s];
        }
        __syncthreads();
    }
    if (tid == 0) {
        float z[5], mx = -1e30f;
        #pragma unroll
        for (int c = 0; c < 5; ++c) {
            float v = red[c][0] + V2b[c];
            v = v > 0.0f ? v : 0.0f;                  // relu before softmax
            z[c] = v;
            mx = v > mx ? v : mx;
        }
        float se = 0.f;
        #pragma unroll
        for (int c = 0; c < 5; ++c) se += __expf(z[c] - mx);
        const float lse = mx + __logf(se);
        #pragma unroll
        for (int c = 0; c < 5; ++c) out[b * 5 + c] = z[c] - lse;
    }
}

// ---------- host-side launch ----------
extern "C" void kernel_launch(void* const* d_in, const int* in_sizes, int n_in,
                              void* d_out, int out_size, void* d_ws, size_t ws_size,
                              hipStream_t stream) {
    (void)in_sizes; (void)n_in; (void)out_size; (void)ws_size;
    constexpr int B = 64, T = 512, D = 1024, H = 2048;

    const float* inputs = (const float*)d_in[0];
    const float* W_w    = (const float*)d_in[1];
    const float* W_b    = (const float*)d_in[2];
    const float* U_w    = (const float*)d_in[3];
    const float* U_b    = (const float*)d_in[4];
    const float* V1_w   = (const float*)d_in[5];
    const float* V1_b   = (const float*)d_in[6];
    const float* V2_w   = (const float*)d_in[7];
    const float* V2_b   = (const float*)d_in[8];

    // workspace carve-up (256B aligned)
    char* ws = (char*)d_ws;
    size_t off = 0;
    auto carve = [&](size_t bytes) -> void* {
        void* p = ws + off;
        off = (off + bytes + 255) & ~(size_t)255;
        return p;
    };
    unsigned short* in_bf = (unsigned short*)carve((size_t)B * T * D * 2);  //  64 MB
    unsigned short* W_bf  = (unsigned short*)carve((size_t)H * D * 2);      //   4 MB
    unsigned short* U_bf  = (unsigned short*)carve((size_t)H * H * 2);      //   8 MB
    unsigned short* V1_bf = (unsigned short*)carve((size_t)H * H * 2);      //   8 MB
    unsigned short* xw_bf = (unsigned short*)carve((size_t)B * T * H * 2);  // 128 MB
    unsigned short* hA    = (unsigned short*)carve((size_t)B * H * 2);      // 256 KB
    unsigned short* hB    = (unsigned short*)carve((size_t)B * H * 2);      // 256 KB
    float*          z1    = (float*)carve((size_t)B * H * 4);               // 512 KB

    // 1) convert operands to bf16
    {
        int n4 = (B * T * D) / 4;
        cvt_f32_bf16_kernel<<<(n4 + 255) / 256, 256, 0, stream>>>(inputs, in_bf, n4);
        n4 = (H * D) / 4;
        cvt_f32_bf16_kernel<<<(n4 + 255) / 256, 256, 0, stream>>>(W_w, W_bf, n4);
        n4 = (H * H) / 4;
        cvt_f32_bf16_kernel<<<(n4 + 255) / 256, 256, 0, stream>>>(U_w, U_bf, n4);
        cvt_f32_bf16_kernel<<<(n4 + 255) / 256, 256, 0, stream>>>(V1_w, V1_bf, n4);
    }
    // 2) h0 = 0
    zero_bf16_kernel<<<(B * H + 255) / 256, 256, 0, stream>>>(hA, B * H);

    // 3) xw = inputs @ W^T + W_b : M = B*T = 32768, N = H, K = D
    {
        dim3 grid((B * T) / 32, H / 256);   // (1024, 8): 32x64 per wave, 4 waves/block
        gemm_xw_kernel<<<grid, 128, 0, stream>>>(in_bf, W_bf, W_b, xw_bf, H, D);
    }

    // 4) recurrence: h = relu(xw[:,t,:] + h @ U^T + U_b), ping-pong hA/hB
    {
        const int nBlocks = H / 64;         // 32 blocks x 4 waves, 64x16 per wave
        for (int t = 0; t < T; ++t) {
            unsigned short* hin  = (t & 1) ? hB : hA;
            unsigned short* hout = (t & 1) ? hA : hB;
            gemm64_relu_kernel<<<nBlocks, 128, 0, stream>>>(
                hin, U_bf, U_b, xw_bf, t, T, hout, nullptr, H, H);
        }
        // T even -> h_last lands in hA
    }

    // 5) z1 = relu(h_last @ V1^T + V1_b)  (fp32 out)
    gemm64_relu_kernel<<<H / 64, 128, 0, stream>>>(
        hA, V1_bf, V1_b, nullptr, 0, T, nullptr, z1, H, H);

    // 6) z2 + log_softmax
    head_kernel<<<B, 256, 0, stream>>>(z1, V2_w, V2_b, (float*)d_out, H);
}
